// MyBertSelfAttention_59846074302608
// MI455X (gfx1250) — matrix operand order repaired and checked by reading, not compiled, not run on hardware
//
#include <hip/hip_runtime.h>
#include <hip/hip_bf16.h>
#include <stdint.h>

// ---------------- CDNA5 WMMA types ----------------
typedef __bf16 v16bf __attribute__((ext_vector_type(16)));
typedef float  v8f   __attribute__((ext_vector_type(8)));

#define Bsz   16
#define Slen  512
#define Hdim  768
#define NHn   12
#define HDn   64
#define NDIST 1023   // 2*MAX_POS-1

__device__ __forceinline__ unsigned short f2bf(float f) {
  union { float f; unsigned int u; } x; x.f = f;
  unsigned int u = x.u;
  u += 0x7FFFu + ((u >> 16) & 1u);   // round-to-nearest-even
  return (unsigned short)(u >> 16);
}

__device__ __forceinline__ v8f wmma_bf16(v16bf a, v16bf b, v8f c) {
  // (neg_a, A, neg_b, B, c_mod, C, reuse_a, reuse_b)
  return __builtin_amdgcn_wmma_f32_16x16x32_bf16(false, a, false, b, (short)0, c,
                                                 false, false);
}

union BF16Frag {           // 16 bf16 per lane = one A or B operand
  uint4 u[2];
  v16bf v;
};

// ---------------- kernel 1: fp32 -> bf16 conversion ----------------
__global__ void cvt_f32_bf16(const float* __restrict__ src,
                             unsigned short* __restrict__ dst, int n) {
  int i = blockIdx.x * blockDim.x + threadIdx.x;
  int stride = gridDim.x * blockDim.x;
  for (; i < n; i += stride) dst[i] = f2bf(src[i]);
}

// ---------------- kernel 2: fused QKV projection (bf16 WMMA GEMM) ----------
// One wave per (16-row M tile, proj p in {q,k,v}, head h). K = 768 (24 steps).
// Q,K stored [bh, s, 64] bf16;  V stored transposed [bh, 64, s] bf16.
__global__ __launch_bounds__(32)
void qkv_gemm(const unsigned short* __restrict__ Xbf,   // [8192,768]
              const unsigned short* __restrict__ Wbf,   // [3][768][768]
              const float* __restrict__ bq,
              const float* __restrict__ bk,
              const float* __restrict__ bv,
              unsigned short* __restrict__ Qbf,         // [192,512,64]
              unsigned short* __restrict__ Kbf,         // [192,512,64]
              unsigned short* __restrict__ Vt)          // [192,64,512]
{
  const int lane = threadIdx.x;
  const int m    = lane & 15;
  const int half = lane >> 4;

  int bid = blockIdx.x;
  int mt  = bid / 36;
  int rem = bid - mt * 36;
  int p   = rem / 12;
  int h   = rem - p * 12;

  const unsigned short* W    = Wbf + (size_t)p * Hdim * Hdim;
  const float*          bias = (p == 0) ? bq : (p == 1) ? bk : bv;

  const int row0 = mt * 16;
  const unsigned short* Xrow = Xbf + (size_t)(row0 + m) * Hdim;

  v8f acc[4] = {};
  for (int c0 = 0; c0 < Hdim; c0 += 32) {
    BF16Frag a;                                    // A: 16x32 of X
    a.u[0] = *(const uint4*)(Xrow + c0 +      8 * half);
    a.u[1] = *(const uint4*)(Xrow + c0 + 16 + 8 * half);
#pragma unroll
    for (int nt = 0; nt < 4; ++nt) {
      int j = h * 64 + nt * 16 + m;                // B[c,j] = W[j,c]
      const unsigned short* Wr = W + (size_t)j * Hdim + c0 + 8 * half;
      BF16Frag b;
      b.u[0] = *(const uint4*)(Wr);
      b.u[1] = *(const uint4*)(Wr + 16);
      acc[nt] = wmma_bf16(a.v, b.v, acc[nt]);
    }
  }

#pragma unroll
  for (int nt = 0; nt < 4; ++nt) {
    int   dcol = nt * 16 + m;                      // d within head
    float bval = bias[h * 64 + dcol];
#pragma unroll
    for (int r = 0; r < 8; ++r) {
      int grow = row0 + r + 8 * half;              // row in [0,8192)
      int bb   = grow >> 9;                        // / 512
      int s    = grow & 511;
      unsigned short v16 = f2bf(acc[nt][r] + bval);
      size_t bh = (size_t)(bb * NHn + h);
      if (p == 0)      Qbf[(bh * Slen + s) * HDn + dcol] = v16;
      else if (p == 1) Kbf[(bh * Slen + s) * HDn + dcol] = v16;
      else             Vt [(bh * HDn + dcol) * Slen + s] = v16;
    }
  }
}

// ---------------- kernel 3: flash attention w/ relative_key_query ----------
// One wave per (b, h, 16-row q tile). Loops k in chunks of 32.
__global__ __launch_bounds__(32)
void attn(const unsigned short* __restrict__ Qbf,   // [192,512,64]
          const unsigned short* __restrict__ Kbf,   // [192,512,64]
          const unsigned short* __restrict__ Vt,    // [192,64,512]
          const unsigned short* __restrict__ Ebf,   // [1023,64]
          const float* __restrict__ mask,           // [B,1,1,S]
          float* __restrict__ out)                  // [B,S,768]
{
  __shared__ float          S_l[16][32];
  __shared__ float          Aq_l[16][48];
  __shared__ float          Ak_l[32][48];
  __shared__ unsigned short P_l[16][32];
  __shared__ float          alpha_l[16];
  __shared__ float          linv_l[16];

  const int lane = threadIdx.x;
  const int m    = lane & 15;
  const int half = lane >> 4;

  int bid = blockIdx.x;
  int qt  = bid & 31;  bid >>= 5;
  int h   = bid % NHn;
  int b   = bid / NHn;
  int bh  = b * NHn + h;
  int q0  = qt * 16;

  const unsigned short* Qbase = Qbf + (size_t)bh * Slen * HDn;
  const unsigned short* Kbase = Kbf + (size_t)bh * Slen * HDn;
  const unsigned short* Vtb   = Vt  + (size_t)bh * HDn * Slen;
  const float*          maskb = mask + (size_t)b * Slen;

  // Q as A-operand fragments, d-chunks [0,32) and [32,64)
  BF16Frag qa[2];
  {
    const unsigned short* Qr = Qbase + (size_t)(q0 + m) * HDn;
    qa[0].u[0] = *(const uint4*)(Qr + 0  + 8 * half);
    qa[0].u[1] = *(const uint4*)(Qr + 16 + 8 * half);
    qa[1].u[0] = *(const uint4*)(Qr + 32 + 8 * half);
    qa[1].u[1] = *(const uint4*)(Qr + 48 + 8 * half);
  }

  v8f   ctx[4] = {};
  float m_i = -3.0e38f, l_i = 0.f;   // per-row state (lanes 0..15)

  for (int k0 = 0; k0 < Slen; k0 += 32) {
    // ---- scores chunk: S[16q x 32k] = Q Kt (2 k-tiles x 2 d-chunks) ----
#pragma unroll
    for (int kt = 0; kt < 2; ++kt) {
      const unsigned short* Kr = Kbase + (size_t)(k0 + kt * 16 + m) * HDn;
      BF16Frag b0, b1;                           // B[d,n] = K[n,d]
      b0.u[0] = *(const uint4*)(Kr + 8 * half);
      b0.u[1] = *(const uint4*)(Kr + 16 + 8 * half);
      b1.u[0] = *(const uint4*)(Kr + 32 + 8 * half);
      b1.u[1] = *(const uint4*)(Kr + 48 + 8 * half);
      v8f s = {};
      s = wmma_bf16(qa[0].v, b0.v, s);
      s = wmma_bf16(qa[1].v, b1.v, s);
#pragma unroll
      for (int r = 0; r < 8; ++r) S_l[r + 8 * half][kt * 16 + m] = s[r];
    }

    // ---- relative-position local GEMMs vs dist_emb slice (48 rows) ----
    // dist(i,j) = (q0+i)-(k0+j)+511, local t = i-j+31 in [0,47]
    int dist0 = q0 - k0 + 480;

    BF16Frag ka[2][2];                           // K chunk rows as A-operand
#pragma unroll
    for (int rt = 0; rt < 2; ++rt) {
      const unsigned short* Kr = Kbase + (size_t)(k0 + rt * 16 + m) * HDn;
      ka[rt][0].u[0] = *(const uint4*)(Kr + 0  + 8 * half);
      ka[rt][0].u[1] = *(const uint4*)(Kr + 16 + 8 * half);
      ka[rt][1].u[0] = *(const uint4*)(Kr + 32 + 8 * half);
      ka[rt][1].u[1] = *(const uint4*)(Kr + 48 + 8 * half);
    }

#pragma unroll
    for (int tt = 0; tt < 3; ++tt) {
      int trow = dist0 + tt * 16 + m;            // clamp OOB (never gathered)
      trow = trow < 0 ? 0 : (trow > 1022 ? 1022 : trow);
      const unsigned short* Er = Ebf + (size_t)trow * HDn;
      BF16Frag e0, e1;                           // B[d,t] = E[t,d]
      e0.u[0] = *(const uint4*)(Er + 8 * half);
      e0.u[1] = *(const uint4*)(Er + 16 + 8 * half);
      e1.u[0] = *(const uint4*)(Er + 32 + 8 * half);
      e1.u[1] = *(const uint4*)(Er + 48 + 8 * half);

      v8f aq = {};
      aq = wmma_bf16(qa[0].v, e0.v, aq);
      aq = wmma_bf16(qa[1].v, e1.v, aq);
#pragma unroll
      for (int r = 0; r < 8; ++r) Aq_l[r + 8 * half][tt * 16 + m] = aq[r];

#pragma unroll
      for (int rt = 0; rt < 2; ++rt) {
        v8f ak = {};
        ak = wmma_bf16(ka[rt][0].v, e0.v, ak);
        ak = wmma_bf16(ka[rt][1].v, e1.v, ak);
#pragma unroll
        for (int r = 0; r < 8; ++r)
          Ak_l[rt * 16 + r + 8 * half][tt * 16 + m] = ak[r];
      }
    }
    __syncthreads();

    // ---- online softmax: lane i owns score row i ----
    if (lane < 16) {
      float sv[32];
      float mc = -3.0e38f;
#pragma unroll
      for (int j = 0; j < 32; ++j) {
        int   t = lane - j + 31;
        float s = S_l[lane][j] + Aq_l[lane][t] + Ak_l[j][t];
        s = s * 0.125f + maskb[k0 + j];          // /sqrt(64) + mask
        sv[j] = s;
        mc = fmaxf(mc, s);
      }
      float mnew = fmaxf(m_i, mc);
      float al   = __expf(m_i - mnew);
      float ls   = 0.f;
#pragma unroll
      for (int j = 0; j < 32; ++j) {
        float p = __expf(sv[j] - mnew);
        ls += p;
        P_l[lane][j] = f2bf(p);
      }
      l_i = l_i * al + ls;
      m_i = mnew;
      alpha_l[lane] = al;
    }
    __syncthreads();

    // ---- ctx += P @ V  (rescale accumulators by alpha per row) ----
    float alv[8];
#pragma unroll
    for (int r = 0; r < 8; ++r) alv[r] = alpha_l[r + 8 * half];

    BF16Frag pa;                                 // P as A-operand from LDS
    pa.u[0] = *(const uint4*)(&P_l[m][8 * half]);
    pa.u[1] = *(const uint4*)(&P_l[m][16 + 8 * half]);

#pragma unroll
    for (int t = 0; t < 4; ++t) {
#pragma unroll
      for (int r = 0; r < 8; ++r) ctx[t][r] *= alv[r];
      // B[k,d] = V[k0+k, 16t+d] = Vt[16t+d][k0+k] (contiguous in k)
      const unsigned short* Vr = Vtb + (size_t)(t * 16 + m) * Slen + k0 + 8 * half;
      BF16Frag vb;
      vb.u[0] = *(const uint4*)(Vr);
      vb.u[1] = *(const uint4*)(Vr + 16);
      ctx[t] = wmma_bf16(pa.v, vb.v, ctx[t]);
    }
    __syncthreads();
  }

  // ---- epilogue: divide by l, store fp32 ----
  if (lane < 16) linv_l[lane] = 1.f / l_i;
  __syncthreads();
  float lv[8];
#pragma unroll
  for (int r = 0; r < 8; ++r) lv[r] = linv_l[r + 8 * half];
#pragma unroll
  for (int t = 0; t < 4; ++t) {
#pragma unroll
    for (int r = 0; r < 8; ++r) {
      int s = q0 + r + 8 * half;
      out[((size_t)b * Slen + s) * Hdim + h * 64 + t * 16 + m] = ctx[t][r] * lv[r];
    }
  }
}

// ---------------- host launch ----------------
extern "C" void kernel_launch(void* const* d_in, const int* in_sizes, int n_in,
                              void* d_out, int out_size, void* d_ws, size_t ws_size,
                              hipStream_t stream) {
  (void)in_sizes; (void)n_in; (void)out_size; (void)ws_size;

  const float* hidden = (const float*)d_in[0];
  const float* mask   = (const float*)d_in[1];
  const float* Wq     = (const float*)d_in[2];
  const float* bq     = (const float*)d_in[3];
  const float* Wk     = (const float*)d_in[4];
  const float* bk     = (const float*)d_in[5];
  const float* Wv     = (const float*)d_in[6];
  const float* bv     = (const float*)d_in[7];
  const float* dist   = (const float*)d_in[8];
  float* out = (float*)d_out;

  // workspace partition (ushort elements)
  const size_t nX = (size_t)Bsz * Slen * Hdim;        // 6,291,456
  const size_t nW = (size_t)Hdim * Hdim;              //   589,824
  const size_t nE = (size_t)NDIST * HDn;              //    65,472
  const size_t nQ = (size_t)Bsz * NHn * Slen * HDn;   // 6,291,456

  unsigned short* Xbf = (unsigned short*)d_ws;
  unsigned short* Wbf = Xbf + nX;                     // 3 matrices
  unsigned short* Ebf = Wbf + 3 * nW;
  unsigned short* Qbf = Ebf + nE;
  unsigned short* Kbf = Qbf + nQ;
  unsigned short* Vt  = Kbf + nQ;

  // 1) convert inputs to bf16
  cvt_f32_bf16<<<2048, 256, 0, stream>>>(hidden, Xbf, (int)nX);
  cvt_f32_bf16<<<1024, 256, 0, stream>>>(Wq, Wbf,          (int)nW);
  cvt_f32_bf16<<<1024, 256, 0, stream>>>(Wk, Wbf + nW,     (int)nW);
  cvt_f32_bf16<<<1024, 256, 0, stream>>>(Wv, Wbf + 2 * nW, (int)nW);
  cvt_f32_bf16<<<256,  256, 0, stream>>>(dist, Ebf, (int)nE);

  // 2) QKV projections: 512 M-tiles x 3 proj x 12 heads, 1 wave each
  qkv_gemm<<<512 * 36, 32, 0, stream>>>(Xbf, Wbf, bq, bk, bv, Qbf, Kbf, Vt);

  // 3) flash attention: (b,h,q-tile) waves
  attn<<<Bsz * NHn * 32, 32, 0, stream>>>(Qbf, Kbf, Vt, Ebf, mask, out);
}
